// DiVeQ_19774029430966
// MI455X (gfx1250) — matrix operand order repaired and checked by reading, not compiled, not run on hardware
//
#include <hip/hip_runtime.h>
#include <hip/hip_bf16.h>
#include <math.h>

typedef _Float16 f16;
typedef _Float16 v16h __attribute__((ext_vector_type(16)));
typedef _Float16 v8h  __attribute__((ext_vector_type(8)));
typedef float    v8f  __attribute__((ext_vector_type(8)));
typedef int      v4i  __attribute__((ext_vector_type(4)));

#define DIMS    128
#define SIGMA_F 0.1f
#define EPS_F   1e-8f

#define STAGE_ROWS 64            // codewords staged per step (4 N-tiles)
#define BSTRIDE    136           // halfs per LDS row (128 + 8 pad: 68-dword
                                 // stride -> each half-wave's ds_load_b128
                                 // covers all 64 banks exactly once)

#if __has_builtin(__builtin_amdgcn_global_load_async_to_lds_b128) && \
    __has_builtin(__builtin_amdgcn_s_wait_asynccnt)
#define USE_ASYNC_LDS 1
typedef __attribute__((address_space(1))) v4i gbl_v4i;   // global 16B payload
typedef __attribute__((address_space(3))) v4i lds_v4i;   // LDS 16B payload
#else
#define USE_ASYNC_LDS 0
#endif

// ---------------------------------------------------------------------------
// Prep: codebook f32 -> f16 (WMMA B-operand), cnorm[k] = ||c_k||^2, zero usage.
// Grid: K blocks x 128 threads.
// ---------------------------------------------------------------------------
__global__ __launch_bounds__(DIMS) void diveq_prep(
    const float* __restrict__ cb, f16* __restrict__ chH,
    float* __restrict__ cnorm, float* __restrict__ usage) {
  const int k = blockIdx.x;
  const int t = threadIdx.x;
  float c = cb[(size_t)k * DIMS + t];
  chH[(size_t)k * DIMS + t] = (f16)c;
  float s = c * c;
  #pragma unroll
  for (int m = 16; m >= 1; m >>= 1) s += __shfl_xor(s, m, 32);
  __shared__ float p[4];
  const int lane = t & 31, wid = t >> 5;
  if (lane == 0) p[wid] = s;
  __syncthreads();
  if (t == 0) {
    cnorm[k] = p[0] + p[1] + p[2] + p[3];
    usage[k] = 0.0f;
  }
}

// ---------------------------------------------------------------------------
// Cooperative stage copy: 64 codewords (halfs 0..63 of each: 64 x 64 halfs = 8KB) from global
// f16 codebook into a padded LDS buffer. 512 16-byte chunks over 256 threads.
// Async path: global_load_async_to_lds_b128 (ASYNCcnt, no VGPR data round
// trip). Fallback: b128 load + ds_store_b128.
// ---------------------------------------------------------------------------
__device__ __forceinline__ void stage_load(const f16* __restrict__ gsrc,
                                           f16* ldst, int tid) {
  #pragma unroll
  for (int i = 0; i < 2; ++i) {
    const int chunk = tid + i * 256;          // 0..511
    const int row   = chunk >> 3;             // 0..63
    const int col   = (chunk & 7) * 8;        // half index within row
    const f16* g = gsrc + (size_t)row * DIMS + col;
    f16*       l = ldst + row * BSTRIDE + col;
#if USE_ASYNC_LDS
    f16* gnc = const_cast<f16*>(g);           // builtin takes non-const v4i*
    __builtin_amdgcn_global_load_async_to_lds_b128(
        (gbl_v4i*)gnc, (lds_v4i*)l, 0, 0);
#else
    *(v8h*)l = *(const v8h*)g;                // global_load_b128 + ds_store_b128
#endif
  }
}

__device__ __forceinline__ void stage_wait() {
#if USE_ASYNC_LDS
  __builtin_amdgcn_s_wait_asynccnt(0);
#endif
  __syncthreads();
}

// ---------------------------------------------------------------------------
// Fused GEMM (z . c^T via v_wmma_f32_16x16x32_f16) + argmin.
// argmin_k d2 == argmax_k (z.c_k - 0.5*||c_k||^2)    (||z||^2 row-constant)
//
// Block = 256 threads = 8 waves; each wave owns a 16-row M tile (128 rows per
// block). All 8 waves share double-buffered LDS stages of 64 codewords, so
// the codebook is read from L2 once per *block* (256MB total) instead of once
// per wave (2GB), while WMMAs for stage s overlap the async DMA of stage s+1.
// Per tile all 8 ds_load_b128s are issued before the 4 chained WMMAs so only
// one s_wait_dscnt is exposed per tile.
//
// VGPR striping per CDNA5 ISA layouts:
//   A 16x32 f16 : lane l row M=l%16; halves K {0..7,16..23} (l<16)
//                                           K {8..15,24..31} (l>=16)
//   B 32x16 f16 : lane l col N=l%16; halves K {0..7,16..23} (l<16) / {8..15,24..31} (l>=16)
//   C/D 16x16   : vgpr j lane l -> (M = j + 8*(l/16), N = l%16)
// ---------------------------------------------------------------------------
__global__ __launch_bounds__(256) void diveq_argmin(
    const float* __restrict__ z, const f16* __restrict__ chH,
    const float* __restrict__ cnorm, float* __restrict__ idxOut, int K) {
  __shared__ f16 bstage[2][STAGE_ROWS * BSTRIDE];   // 2 x 17KB

  const int tid   = threadIdx.x;
  const int lane  = tid & 31;
  const int wid   = tid >> 5;                  // 0..7
  const int nlane = lane & 15;
  const int hi    = lane >> 4;                 // 0/1 half-wave
  const int m0    = (blockIdx.x * 8 + wid) * 16;

  // ---- A fragments: load + convert once, reused for all K/16 N-tiles ----
  const float* zr = z + (size_t)(m0 + nlane) * DIMS;
  v16h a[4];
  #pragma unroll
  for (int c = 0; c < 4; ++c) {
    const int base = c * 32 + hi * 8;
    #pragma unroll
    for (int i = 0; i < 8; ++i) a[c][i]     = (f16)zr[base + i];
    #pragma unroll
    for (int i = 0; i < 8; ++i) a[c][8 + i] = (f16)zr[base + 16 + i];
  }

  float best[8];
  int   bidx[8];
  #pragma unroll
  for (int j = 0; j < 8; ++j) { best[j] = -3.4e38f; bidx[j] = 0; }

  const int NS = K / STAGE_ROWS;               // 64 stages
  stage_load(chH, &bstage[0][0], tid);         // prologue: stage 0
  stage_wait();

  for (int s = 0; s < NS; ++s) {
    f16* cur = &bstage[s & 1][0];
    if (s + 1 < NS)                            // DMA next stage while computing
      stage_load(chH + (size_t)(s + 1) * STAGE_ROWS * DIMS,
                 &bstage[(s + 1) & 1][0], tid);

    #pragma unroll
    for (int t = 0; t < 4; ++t) {              // 4 N-tiles of 16 codewords
      const f16* brow = cur + (t * 16 + nlane) * BSTRIDE + hi * 8;

      // Issue all 8 LDS fragment loads first (distinct registers), then run
      // the 4 accumulator-chained WMMAs behind a single dscnt wait.
      union { v16h v; v8h h[2]; } bf[4];
      #pragma unroll
      for (int c = 0; c < 4; ++c) {            // K = 4 x 32
        bf[c].h[0] = *(const v8h*)(brow + c * 32);
        bf[c].h[1] = *(const v8h*)(brow + c * 32 + 16);
      }
      v8f acc = {0.f, 0.f, 0.f, 0.f, 0.f, 0.f, 0.f, 0.f};
      #pragma unroll
      for (int c = 0; c < 4; ++c)
        acc = __builtin_amdgcn_wmma_f32_16x16x32_f16(
                  false, a[c], false, bf[c].v, (short)0, acc, false, false);

      const int   idx = s * STAGE_ROWS + t * 16 + nlane;
      const float sadj = -0.5f * cnorm[idx];
      #pragma unroll
      for (int j = 0; j < 8; ++j) {
        float sc = acc[j] + sadj;              // larger == smaller d2
        if (sc > best[j]) { best[j] = sc; bidx[j] = idx; }  // strict > keeps
      }                                                     // earliest (argmin)
    }
    stage_wait();                              // next stage landed; cur free
  }

  // Reduce across the 16 lanes holding the same rows (masks < 16 stay inside
  // each half-wave). Tie-break to the lower index (first min).
  #pragma unroll
  for (int m = 1; m < 16; m <<= 1) {
    #pragma unroll
    for (int j = 0; j < 8; ++j) {
      float os = __shfl_xor(best[j], m, 32);
      int   oi = __shfl_xor(bidx[j], m, 32);
      if (os > best[j] || (os == best[j] && oi < bidx[j])) {
        best[j] = os; bidx[j] = oi;
      }
    }
  }
  if (nlane == 0) {
    #pragma unroll
    for (int j = 0; j < 8; ++j)
      idxOut[m0 + hi * 8 + j] = (float)bidx[j];
  }
}

// ---------------------------------------------------------------------------
// Finalize: d = c[idx]-z, dist=||d||, v_d = v*sigma + d,
// z_q = z + dist * v_d/(||v_d||+eps); usage scatter; stash d2 for commit loss.
// Grid: B blocks x 128 threads.
// ---------------------------------------------------------------------------
__global__ __launch_bounds__(DIMS) void diveq_finalize(
    const float* __restrict__ z, const float* __restrict__ v,
    const float* __restrict__ cb, const float* __restrict__ idxF,
    float* __restrict__ zq, float* __restrict__ distOut,
    float* __restrict__ usage, float* __restrict__ dist2ws) {
  const int b = blockIdx.x, t = threadIdx.x;
  const int idx = (int)idxF[b];
  const float zc = z[(size_t)b * DIMS + t];
  const float nr = cb[(size_t)idx * DIMS + t];
  const float dv = nr - zc;
  const float vd = v[(size_t)b * DIMS + t] * SIGMA_F + dv;

  float s1 = dv * dv, s2 = vd * vd;
  #pragma unroll
  for (int m = 16; m >= 1; m >>= 1) {
    s1 += __shfl_xor(s1, m, 32);
    s2 += __shfl_xor(s2, m, 32);
  }
  __shared__ float p1[4], p2[4];
  const int lane = t & 31, wid = t >> 5;
  if (lane == 0) { p1[wid] = s1; p2[wid] = s2; }
  __syncthreads();
  const float dsum = p1[0] + p1[1] + p1[2] + p1[3];
  const float vsum = p2[0] + p2[1] + p2[2] + p2[3];
  const float dist  = sqrtf(dsum);
  const float scale = dist / (sqrtf(vsum) + EPS_F);
  zq[(size_t)b * DIMS + t] = zc + scale * vd;
  if (t == 0) {
    distOut[b] = dist;
    dist2ws[b] = dsum;
    atomicAdd(&usage[idx], 1.0f);   // sums of 1.0f: exact & order-free
  }
}

// ---------------------------------------------------------------------------
// Commit loss: deterministic fixed-order tree reduction of per-row d2.
// ---------------------------------------------------------------------------
__global__ __launch_bounds__(1024) void diveq_commit(
    const float* __restrict__ dist2, float* __restrict__ commitOut, int B) {
  __shared__ float part[32];
  const int t = threadIdx.x;
  float s = 0.f;
  for (int i = t; i < B; i += 1024) s += dist2[i];
  #pragma unroll
  for (int m = 16; m >= 1; m >>= 1) s += __shfl_xor(s, m, 32);
  const int lane = t & 31, wid = t >> 5;
  if (lane == 0) part[wid] = s;
  __syncthreads();
  if (t < 32) {
    s = part[t];
    #pragma unroll
    for (int m = 16; m >= 1; m >>= 1) s += __shfl_xor(s, m, 32);
    if (t == 0) commitOut[0] = s / ((float)B * (float)DIMS);
  }
}

// ---------------------------------------------------------------------------
extern "C" void kernel_launch(void* const* d_in, const int* in_sizes, int n_in,
                              void* d_out, int out_size, void* d_ws, size_t ws_size,
                              hipStream_t stream) {
  const float* z  = (const float*)d_in[0];
  const float* v  = (const float*)d_in[1];
  const float* cb = (const float*)d_in[2];
  const int B = in_sizes[0] / DIMS;   // 32768
  const int K = in_sizes[2] / DIMS;   // 4096

  // d_out layout (reference return order, flat f32):
  //   z_q[B*D] | indices[B] | dist[B] | commit_loss[1] | usage_count[K]
  float* out       = (float*)d_out;
  float* zq        = out;
  float* idxOut    = out + (size_t)B * DIMS;
  float* distOut   = idxOut + B;
  float* commitOut = distOut + B;
  float* usage     = commitOut + 1;

  // Workspace: cnorm[K] f32 | dist2[B] f32 | codebook f16 [K*D] (32B-aligned)
  char*  ws    = (char*)d_ws;
  float* cnorm = (float*)ws;
  float* dist2 = (float*)(ws + (size_t)K * 4);
  f16*   chH   = (f16*)(ws + (size_t)K * 4 + (size_t)B * 4);

  diveq_prep    <<<K,       DIMS, 0, stream>>>(cb, chH, cnorm, usage);
  diveq_argmin  <<<B / 128, 256,  0, stream>>>(z, chH, cnorm, idxOut, K);
  diveq_finalize<<<B,       DIMS, 0, stream>>>(z, v, cb, idxOut, zq, distOut,
                                               usage, dist2);
  diveq_commit  <<<1,       1024, 0, stream>>>(dist2, commitOut, B);
}
